// SpikingGooseRecurrentLayer_37761352466564
// MI455X (gfx1250) — matrix operand order repaired and checked by reading, not compiled, not run on hardware
//
#include <hip/hip_runtime.h>
#include <math.h>

#define D_MODEL 1024
#define BATCH   16
#define SEQ     1024
#define M_TOTAL (BATCH * SEQ)        // 16384 rows (b,t flattened)
#define THRESH  0.5f

typedef __attribute__((ext_vector_type(16))) __bf16 v16bf;
typedef __attribute__((ext_vector_type(8)))  __bf16 v8bf;
typedef __attribute__((ext_vector_type(8)))  float  v8f;

typedef __attribute__((ext_vector_type(4))) unsigned u32x4;
typedef __attribute__((ext_vector_type(8))) int      i32x8;
typedef __attribute__((ext_vector_type(4))) int      i32x4;

#if __has_builtin(__builtin_amdgcn_tensor_load_to_lds)
#define USE_TDM 1
#else
#define USE_TDM 0
#endif

// ---------------------------------------------------------------------------
// LDS addressing + CDNA5 async copy (fallback path, ASYNCcnt-tracked)
// ---------------------------------------------------------------------------
__device__ __forceinline__ unsigned lds_off_u32(const void* p) {
    // AS3 pointer value == byte offset within the wave's LDS allocation
    return (unsigned)(size_t)(__attribute__((address_space(3))) const void*)p;
}

__device__ __forceinline__ void async_load_b128(unsigned lds_addr,
                                                const void* gptr) {
    asm volatile("global_load_async_to_lds_b128 %0, %1, off"
                 :: "v"(lds_addr),
                    "v"((unsigned long long)(size_t)gptr)
                 : "memory");
}

__device__ __forceinline__ void wait_async() {
    asm volatile("s_wait_asynccnt 0x0" ::: "memory");
}

// ---------------------------------------------------------------------------
// TDM: one descriptor moves a full 128(col) x 32(K) bf16 weight tile to LDS.
// D# packing per CDNA5 ISA (2D, no padding,
// no gather, no multicast). LDS fill order = [col][k] contiguous in k,
// which is exactly the layout load_b_frag_lds() consumes.
// ---------------------------------------------------------------------------
#if USE_TDM
__device__ __forceinline__ void tdm_load_w_tile(const __bf16* W,
                                                unsigned lds_byte_off,
                                                int col0, int k0) {
    const unsigned long long ga =
        (unsigned long long)(size_t)(W + (size_t)col0 * D_MODEL + k0);

    u32x4 g0;
    g0[0] = 1u;                                   // count=1 valid user D#
    g0[1] = lds_byte_off;                         // lds_addr [63:32]
    g0[2] = (unsigned)(ga & 0xFFFFFFFFu);         // global_addr[31:0]
    g0[3] = (unsigned)((ga >> 32) & 0x01FFFFFFu)  // global_addr[56:32]
          | (2u << 30);                           // type = 2 ("image")

    i32x8 g1;
    g1[0] = 1 << 16;            // wg_mask=0, data_size=1 (2 bytes)
    g1[1] = 0x400 << 16;        // tensor_dim0[15:0]=1024 @ bits[63:48]
    g1[2] = 0x400 << 16;        // tensor_dim0 hi=0; tensor_dim1[15:0]=1024
    g1[3] = 32 << 16;           // tensor_dim1 hi=0; tile_dim0=32 (K)
    g1[4] = 128;                // tile_dim1=128 (cols); tile_dim2=0
    g1[5] = 1024;               // tensor_dim0_stride[31:0] = 1024 elems
    g1[6] = 0;                  // stride hi, dim1_stride lo
    g1[7] = 0;

    const i32x4 z4 = {0, 0, 0, 0};
#if defined(__clang_major__) && (__clang_major__ >= 23)
    const i32x8 z8 = {0, 0, 0, 0, 0, 0, 0, 0};
    __builtin_amdgcn_tensor_load_to_lds(g0, g1, z4, z4, z8, 0);
#else
    __builtin_amdgcn_tensor_load_to_lds(g0, g1, z4, z4, 0);
#endif
}
#endif

// ---------------------------------------------------------------------------
// WMMA helper: D = A(16x32 bf16) * B(32x16 bf16) + C(16x16 f32)
// ---------------------------------------------------------------------------
__device__ __forceinline__ v8f wmma_bf16(v16bf a, v16bf b, v8f c) {
    return __builtin_amdgcn_wmma_f32_16x16x32_bf16(
        /*neg_a=*/false, a, /*neg_b=*/false, b,
        /*c_mod=*/(short)0, c, /*reuse_a=*/false, /*reuse_b=*/false);
}

// A fragment (16x32, bf16). contiguous k order used here (matches B):
//   lanes 0-15  : row M=lane,    VGPR0-7 = K 0..15
//   lanes 16-31 : row M=lane-16, VGPR0-7 = K 16..31
__device__ __forceinline__ v16bf load_a_frag(const __bf16* __restrict__ A,
                                             int row, int k0, int grp) {
    const __bf16* p = A + (size_t)row * D_MODEL + k0 + (grp << 4);
    union { v16bf v; v8bf h[2]; } u;
    u.h[0] = *(const v8bf*)(p);
    u.h[1] = *(const v8bf*)(p + 8);
    return u.v;
}

// B fragment from LDS-staged weight tile ([colLocal][kk], kk contiguous).
//   lanes 0-15  : col N=lane,    K = k0..k0+15
//   lanes 16-31 : col N=lane-16, K = k0+16..k0+31
__device__ __forceinline__ v16bf load_b_frag_lds(const __bf16* smem,
                                                 int colLocal, int grp) {
    return *(const v16bf*)(smem + colLocal * 32 + (grp << 4));
}

__device__ __forceinline__ float ternary(float x) {
    return x > THRESH ? 1.0f : (x < -THRESH ? -1.0f : 0.0f);
}
__device__ __forceinline__ float sigmoidf(float x) {
    return 1.0f / (1.0f + __expf(-x));
}

// ---------------------------------------------------------------------------
// 1) LayerNorm: one block (256 thr) per row of 1024 elems
// ---------------------------------------------------------------------------
__global__ void __launch_bounds__(256)
ln_kernel(const float* __restrict__ x, const float* __restrict__ gamma,
          const float* __restrict__ beta, float* __restrict__ xnorm) {
    __shared__ float red[256];
    const int row = blockIdx.x;
    const int tid = threadIdx.x;
    const float4 v = ((const float4*)(x + (size_t)row * D_MODEL))[tid];

    red[tid] = v.x + v.y + v.z + v.w;
    __syncthreads();
    for (int off = 128; off > 0; off >>= 1) {
        if (tid < off) red[tid] += red[tid + off];
        __syncthreads();
    }
    const float mu = red[0] * (1.0f / D_MODEL);
    __syncthreads();

    const float d0 = v.x - mu, d1 = v.y - mu, d2 = v.z - mu, d3 = v.w - mu;
    red[tid] = d0 * d0 + d1 * d1 + d2 * d2 + d3 * d3;
    __syncthreads();
    for (int off = 128; off > 0; off >>= 1) {
        if (tid < off) red[tid] += red[tid + off];
        __syncthreads();
    }
    const float inv = rsqrtf(red[0] * (1.0f / D_MODEL) + 1e-5f);

    const float4 g = ((const float4*)gamma)[tid];
    const float4 b = ((const float4*)beta)[tid];
    float4 o;
    o.x = d0 * inv * g.x + b.x;
    o.y = d1 * inv * g.y + b.y;
    o.z = d2 * inv * g.z + b.z;
    o.w = d3 * inv * g.w + b.w;
    ((float4*)(xnorm + (size_t)row * D_MODEL))[tid] = o;
}

// ---------------------------------------------------------------------------
// 2) Time-mix: xk/xv/xr = xn*mix + prev*(1-mix), emit bf16
// ---------------------------------------------------------------------------
__global__ void __launch_bounds__(256)
mix_kernel(const float* __restrict__ xnorm,
           const float* __restrict__ mk, const float* __restrict__ mv,
           const float* __restrict__ mr,
           __bf16* __restrict__ xk, __bf16* __restrict__ xv,
           __bf16* __restrict__ xr) {
    const size_t idx = (size_t)blockIdx.x * blockDim.x + threadIdx.x;
    const int d = (int)(idx & (D_MODEL - 1));
    const int t = (int)((idx >> 10) & (SEQ - 1));
    const float xn = xnorm[idx];
    const float pv = (t == 0) ? 0.0f : xnorm[idx - D_MODEL];
    float a;
    a = mk[d]; xk[idx] = (__bf16)(xn * a + pv * (1.0f - a));
    a = mv[d]; xv[idx] = (__bf16)(xn * a + pv * (1.0f - a));
    a = mr[d]; xr[idx] = (__bf16)(xn * a + pv * (1.0f - a));
}

// ---------------------------------------------------------------------------
// 3) f32 -> bf16 weight conversion
// ---------------------------------------------------------------------------
__global__ void __launch_bounds__(256)
cvt_bf16_kernel(const float* __restrict__ src, __bf16* __restrict__ dst, int n) {
    const int idx = blockIdx.x * blockDim.x + threadIdx.x;
    if (idx < n) dst[idx] = (__bf16)src[idx];
}

// ---------------------------------------------------------------------------
// GEMM geometry: block = 8 waves (4M x 2N), block tile 128M x 128N,
// wave tile 32M x 64N (2x4 WMMA subtiles), K-step 32.
// Weight tiles staged to LDS: TDM descriptor (1 instr/tile, TENSORcnt) when
// available, else per-lane async copies (ASYNCcnt). Double-buffered.
// ---------------------------------------------------------------------------
#define BTILE_ELEMS (128 * 32)   // one weight tile: 128 cols x 32 K

__device__ __forceinline__ void stage_w_tile_async(const __bf16* __restrict__ W,
                                                   __bf16* smem, int tid,
                                                   int nBlkBase, int k0) {
    const int c    = tid & 127;      // local col
    const int half = tid >> 7;       // 0/1 -> K 0..15 / 16..31
    const __bf16* src = W + (size_t)(nBlkBase + c) * D_MODEL + k0 + (half << 4);
    const unsigned dst = lds_off_u32(smem + c * 32 + (half << 4));
    async_load_b128(dst, src);
    async_load_b128(dst + 16, src + 8);
}

// Stage both weight tiles for one K-slice.
__device__ __forceinline__ void stage_pair(const __bf16* __restrict__ w0,
                                           const __bf16* __restrict__ w1,
                                           __bf16* sm0, __bf16* sm1,
                                           int tid, int wave,
                                           int nBlkBase, int k0) {
#if USE_TDM
    if (wave == 0) {
        tdm_load_w_tile(w0, lds_off_u32(sm0), nBlkBase, k0);
        tdm_load_w_tile(w1, lds_off_u32(sm1), nBlkBase, k0);
    }
    (void)tid;
#else
    stage_w_tile_async(w0, sm0, tid, nBlkBase, k0);
    stage_w_tile_async(w1, sm1, tid, nBlkBase, k0);
#endif
}

// Drain staging engine, then block barrier (release staged LDS to all waves).
__device__ __forceinline__ void drain_stage(int wave) {
#if USE_TDM
    if (wave == 0) __builtin_amdgcn_s_wait_tensorcnt((short)0);
#else
    (void)wave;
    wait_async();
#endif
    __syncthreads();
}

// ---------------------------------------------------------------------------
// 4) Fused k/v GEMM (WMMA bf16) + ternary spike + kv product
// ---------------------------------------------------------------------------
__global__ void __launch_bounds__(256)
gemm_kv_kernel(const __bf16* __restrict__ xk, const __bf16* __restrict__ xv,
               const __bf16* __restrict__ wk, const __bf16* __restrict__ wv,
               float* __restrict__ kv) {
    __shared__ __bf16 smK[2][BTILE_ELEMS];
    __shared__ __bf16 smV[2][BTILE_ELEMS];

    const int tid  = threadIdx.x;
    const int lane = tid & 31;
    const int wave = tid >> 5;
    const int mBlk = blockIdx.x & 127;           // 128 M-blocks of 128 rows
    const int nBlk = blockIdx.x >> 7;            // 8 N-blocks of 128 cols
    const int nBlkBase = nBlk << 7;
    const int waveM = (mBlk << 7) + ((wave & 3) << 5);   // wave 32-row base
    const int nwLoc = (wave >> 2) << 6;                  // wave 64-col base (local)
    const int grp = lane >> 4;
    const int ln  = lane & 15;

    v8f accK[2][4] = {}, accV[2][4] = {};

    stage_pair(wk, wv, smK[0], smV[0], tid, wave, nBlkBase, 0);
    drain_stage(wave);

    int buf = 0;
    for (int k0 = 0; k0 < D_MODEL; k0 += 32) {
        if (k0 + 32 < D_MODEL)
            stage_pair(wk, wv, smK[buf ^ 1], smV[buf ^ 1], tid, wave,
                       nBlkBase, k0 + 32);

        v16bf aK[2], aV[2];
#pragma unroll
        for (int mi = 0; mi < 2; ++mi) {
            aK[mi] = load_a_frag(xk, waveM + mi * 16 + ln, k0, grp);
            aV[mi] = load_a_frag(xv, waveM + mi * 16 + ln, k0, grp);
        }

#pragma unroll
        for (int j = 0; j < 4; ++j) {
            const int colLocal = nwLoc + j * 16 + ln;
            const v16bf bK = load_b_frag_lds(smK[buf], colLocal, grp);
            const v16bf bV = load_b_frag_lds(smV[buf], colLocal, grp);
#pragma unroll
            for (int mi = 0; mi < 2; ++mi) {
                accK[mi][j] = wmma_bf16(aK[mi], bK, accK[mi][j]);
                accV[mi][j] = wmma_bf16(aV[mi], bV, accV[mi][j]);
            }
        }

        drain_stage(wave);
        buf ^= 1;
    }

    // Epilogue: ternary(k)*ternary(v) -> kv
#pragma unroll
    for (int mi = 0; mi < 2; ++mi) {
        const int rowOff = waveM + mi * 16 + (grp << 3);
#pragma unroll
        for (int j = 0; j < 4; ++j) {
            const int col = nBlkBase + nwLoc + j * 16 + ln;
#pragma unroll
            for (int i = 0; i < 8; ++i) {
                kv[(size_t)(rowOff + i) * D_MODEL + col] =
                    ternary(accK[mi][j][i]) * ternary(accV[mi][j][i]);
            }
        }
    }
}

// ---------------------------------------------------------------------------
// 5) Sequential scan: S[t] = sigmoid(decay_w)*S[t-1] + kv[t], per (b,d)
// ---------------------------------------------------------------------------
__global__ void __launch_bounds__(256)
scan_kernel(const float* __restrict__ kv, const float* __restrict__ S0,
            const float* __restrict__ decay_w,
            __bf16* __restrict__ s_all, float* __restrict__ s_final) {
    const int idx = blockIdx.x * blockDim.x + threadIdx.x;   // 0..B*D-1
    const int d = idx & (D_MODEL - 1);
    const int b = idx >> 10;
    const float decay = sigmoidf(decay_w[d]);
    float S = S0[idx];
    const float* kvp = kv + (size_t)b * SEQ * D_MODEL + d;
    __bf16* sp = s_all + (size_t)b * SEQ * D_MODEL + d;
    for (int t = 0; t < SEQ; ++t) {
        S = decay * S + kvp[(size_t)t * D_MODEL];
        sp[(size_t)t * D_MODEL] = (__bf16)S;
    }
    s_final[idx] = S;
}

// ---------------------------------------------------------------------------
// 6) Fused r GEMM + output GEMM: out = x + sigmoid(xr@Wr^T) * (S_all@Wo^T)
// ---------------------------------------------------------------------------
__global__ void __launch_bounds__(256)
gemm_out_kernel(const __bf16* __restrict__ xr, const __bf16* __restrict__ s_all,
                const __bf16* __restrict__ wr, const __bf16* __restrict__ wo,
                const float* __restrict__ x, float* __restrict__ out) {
    __shared__ __bf16 smR[2][BTILE_ELEMS];
    __shared__ __bf16 smO[2][BTILE_ELEMS];

    const int tid  = threadIdx.x;
    const int lane = tid & 31;
    const int wave = tid >> 5;
    const int mBlk = blockIdx.x & 127;
    const int nBlk = blockIdx.x >> 7;
    const int nBlkBase = nBlk << 7;
    const int waveM = (mBlk << 7) + ((wave & 3) << 5);
    const int nwLoc = (wave >> 2) << 6;
    const int grp = lane >> 4;
    const int ln  = lane & 15;

    v8f accR[2][4] = {}, accO[2][4] = {};

    stage_pair(wr, wo, smR[0], smO[0], tid, wave, nBlkBase, 0);
    drain_stage(wave);

    int buf = 0;
    for (int k0 = 0; k0 < D_MODEL; k0 += 32) {
        if (k0 + 32 < D_MODEL)
            stage_pair(wr, wo, smR[buf ^ 1], smO[buf ^ 1], tid, wave,
                       nBlkBase, k0 + 32);

        v16bf aR[2], aS[2];
#pragma unroll
        for (int mi = 0; mi < 2; ++mi) {
            aR[mi] = load_a_frag(xr,    waveM + mi * 16 + ln, k0, grp);
            aS[mi] = load_a_frag(s_all, waveM + mi * 16 + ln, k0, grp);
        }

#pragma unroll
        for (int j = 0; j < 4; ++j) {
            const int colLocal = nwLoc + j * 16 + ln;
            const v16bf bR = load_b_frag_lds(smR[buf], colLocal, grp);
            const v16bf bO = load_b_frag_lds(smO[buf], colLocal, grp);
#pragma unroll
            for (int mi = 0; mi < 2; ++mi) {
                accR[mi][j] = wmma_bf16(aR[mi], bR, accR[mi][j]);
                accO[mi][j] = wmma_bf16(aS[mi], bO, accO[mi][j]);
            }
        }

        drain_stage(wave);
        buf ^= 1;
    }

#pragma unroll
    for (int mi = 0; mi < 2; ++mi) {
        const int rowOff = waveM + mi * 16 + (grp << 3);
#pragma unroll
        for (int j = 0; j < 4; ++j) {
            const int col = nBlkBase + nwLoc + j * 16 + ln;
#pragma unroll
            for (int i = 0; i < 8; ++i) {
                const size_t o = (size_t)(rowOff + i) * D_MODEL + col;
                out[o] = x[o] + sigmoidf(accR[mi][j][i]) * accO[mi][j][i];
            }
        }
    }
}

// ---------------------------------------------------------------------------
// Host-side launcher
// ---------------------------------------------------------------------------
extern "C" void kernel_launch(void* const* d_in, const int* in_sizes, int n_in,
                              void* d_out, int out_size, void* d_ws, size_t ws_size,
                              hipStream_t stream) {
    const float* x        = (const float*)d_in[0];
    const float* S0       = (const float*)d_in[1];
    const float* ln_gamma = (const float*)d_in[2];
    const float* ln_beta  = (const float*)d_in[3];
    const float* mix_k    = (const float*)d_in[4];
    const float* mix_v    = (const float*)d_in[5];
    const float* mix_r    = (const float*)d_in[6];
    const float* decay_w  = (const float*)d_in[7];
    const float* Wk       = (const float*)d_in[8];
    const float* Wv       = (const float*)d_in[9];
    const float* Wr       = (const float*)d_in[10];
    const float* Wo       = (const float*)d_in[11];

    float* out     = (float*)d_out;                       // (B,T,D)
    float* s_final = out + (size_t)M_TOTAL * D_MODEL;     // (B,D)

    char* ws = (char*)d_ws;
    size_t off = 0;
    float*  xnorm = (float*)(ws + off);  off += (size_t)M_TOTAL * D_MODEL * 4;
    __bf16* xk    = (__bf16*)(ws + off); off += (size_t)M_TOTAL * D_MODEL * 2;
    __bf16* xv    = (__bf16*)(ws + off); off += (size_t)M_TOTAL * D_MODEL * 2;
    __bf16* xr    = (__bf16*)(ws + off); off += (size_t)M_TOTAL * D_MODEL * 2;
    __bf16* wkb   = (__bf16*)(ws + off); off += (size_t)D_MODEL * D_MODEL * 2;
    __bf16* wvb   = (__bf16*)(ws + off); off += (size_t)D_MODEL * D_MODEL * 2;
    __bf16* wrb   = (__bf16*)(ws + off); off += (size_t)D_MODEL * D_MODEL * 2;
    __bf16* wob   = (__bf16*)(ws + off); off += (size_t)D_MODEL * D_MODEL * 2;
    float*  kv    = (float*)(ws + off);  off += (size_t)M_TOTAL * D_MODEL * 4;
    __bf16* s_all = (__bf16*)(ws + off); off += (size_t)M_TOTAL * D_MODEL * 2;
    (void)ws_size; (void)n_in; (void)in_sizes; (void)out_size;

    const int W_ELEMS = D_MODEL * D_MODEL;

    // 1) LayerNorm
    ln_kernel<<<M_TOTAL, 256, 0, stream>>>(x, ln_gamma, ln_beta, xnorm);

    // 2) Time-mix -> bf16
    mix_kernel<<<(M_TOTAL * D_MODEL) / 256, 256, 0, stream>>>(
        xnorm, mix_k, mix_v, mix_r, xk, xv, xr);

    // 3) Weights -> bf16
    cvt_bf16_kernel<<<W_ELEMS / 256, 256, 0, stream>>>(Wk, wkb, W_ELEMS);
    cvt_bf16_kernel<<<W_ELEMS / 256, 256, 0, stream>>>(Wv, wvb, W_ELEMS);
    cvt_bf16_kernel<<<W_ELEMS / 256, 256, 0, stream>>>(Wr, wrb, W_ELEMS);
    cvt_bf16_kernel<<<W_ELEMS / 256, 256, 0, stream>>>(Wo, wob, W_ELEMS);

    // 4) k/v WMMA GEMMs + ternary + kv product: 128 x 8 block tiles
    gemm_kv_kernel<<<1024, 256, 0, stream>>>(xk, xv, wkb, wvb, kv);

    // 5) Recurrent scan
    scan_kernel<<<(BATCH * D_MODEL) / 256, 256, 0, stream>>>(
        kv, S0, decay_w, s_all, s_final);

    // 6) r GEMM + output GEMM + residual
    gemm_out_kernel<<<1024, 256, 0, stream>>>(xr, s_all, wrb, wob, x, out);
}
